// ISGN_18786186952881
// MI455X (gfx1250) — matrix operand order repaired and checked by reading, not compiled, not run on hardware
//
#include <hip/hip_runtime.h>
#include <math.h>

// ---------------------------------------------------------------------------
// GatedGraph GNN for MI455X (gfx1250, wave32).
// All GEMMs via v_wmma_f32_16x16x32_bf16 (fp32 accum). All operand tensors are
// pre-transposed/pre-converted once per call so every LDS tile is a straight
// 2D copy -> staged by the Tensor Data Mover (tensor_load_to_lds) with native
// LDS padding (row stride 40 halfwords), double-buffered and synchronized via
// s_wait_tensorcnt. Fallback to vectorized b128 staging if TDM builtin absent.
// em (bf16, 80MB) stays resident in the 192MB L2 across all 10 iterations.
// ---------------------------------------------------------------------------

#ifndef __has_builtin
#define __has_builtin(x) 0
#endif

#if __has_builtin(__builtin_amdgcn_tensor_load_to_lds) && \
    __has_builtin(__builtin_amdgcn_s_wait_tensorcnt)
#define HAVE_TDM 1
#else
#define HAVE_TDM 0
#endif

typedef __bf16 bf16_t;
typedef bf16_t v16bf __attribute__((ext_vector_type(16)));
typedef float  v8f   __attribute__((ext_vector_type(8)));

constexpr int N_ = 2048;   // nodes
constexpr int S_ = 256;    // hidden size
constexpr int E_ = 10;     // edge types
constexpr int ITER_ = 10;  // reference scan length (python scalar, fixed)

constexpr int BM = 128;    // output rows per block
constexpr int BN = 128;    // output cols per block
constexpr int BK = 32;     // k-step (one 16x16x32 WMMA deep)
constexpr int LDW = 40;    // LDS row stride in halfwords (32 data + 8 pad)

__device__ __forceinline__ unsigned short f2bf(float f) {
    unsigned int u = __float_as_uint(f);
    u += 0x7fffu + ((u >> 16) & 1u);   // round-to-nearest-even
    return (unsigned short)(u >> 16);
}

union FragU { uint4 q[2]; v16bf v; };

// A fragment (16x32 bf16): lane l -> row = l&15, K chunks {kh*8..+7, 16+kh*8..+7}
__device__ __forceinline__ v16bf ld_fragA(const unsigned short* lds, int row, int kh) {
    FragU f;
    const uint4* p = (const uint4*)(lds + row * LDW + kh * 8);
    f.q[0] = p[0];
    f.q[1] = p[2];
    return f.v;
}

// B fragment (32x16 bf16): lane l -> col = l&15, K chunks {kh*8..+7, 16+kh*8..+7}
__device__ __forceinline__ v16bf ld_fragB(const unsigned short* lds, int col, int kh) {
    FragU f;
    const uint4* p = (const uint4*)(lds + col * LDW + kh * 8);
    f.q[0] = p[0];
    f.q[1] = p[2];
    return f.v;
}

__device__ __forceinline__ v8f v8f_zero() {
    v8f z = {0.f, 0.f, 0.f, 0.f, 0.f, 0.f, 0.f, 0.f};
    return z;
}

#if HAVE_TDM
typedef unsigned int u32x4 __attribute__((ext_vector_type(4)));
typedef int          i32x4 __attribute__((ext_vector_type(4)));
typedef int          i32x8 __attribute__((ext_vector_type(8)));

// TDM: copy a 128-row x 32-halfword 2D tile (row stride = strideElems) from
// global to LDS, inserting 8 halfwords of pad per row (pad_interval = 16
// DWORDs, pad_amount = 4 DWORDs) -> LDS row stride 40 = LDW.
__device__ __forceinline__ void tdm_tile_128x32(unsigned ldsOff,
                                                const unsigned short* g,
                                                unsigned strideElems) {
    unsigned long long ga = (unsigned long long)(size_t)g;
    u32x4 g0;
    g0.x = 1u;                                                  // count=1 (valid user D#)
    g0.y = ldsOff;                                              // lds_addr
    g0.z = (unsigned)ga;                                        // global_addr[31:0]
    g0.w = (unsigned)((ga >> 32) & 0x1ffffffull) | (2u << 30);  // addr[56:32] | type=2
    i32x8 g1;
    g1[0] = (1 << 16)      // data_size = 1 -> 2 bytes
          | (1 << 20)      // pad_enable
          | (3 << 22)      // pad_interval: 16 DWORDs (= one 32-halfword row)
          | (3 << 25);     // pad_amount:   4 DWORDs (= 8 halfwords)
    g1[1] = (32 & 0xffff) << 16;        // tensor_dim0 = 32 (tile == tensor view)
    g1[2] = (128 & 0xffff) << 16;       // tensor_dim0 hi = 0 | tensor_dim1 = 128
    g1[3] = (32 << 16);                 // tensor_dim1 hi = 0 | tile_dim0 = 32
    g1[4] = 128;                        // tile_dim1 = 128, tile_dim2 = 0
    g1[5] = (int)strideElems;           // tensor_dim0_stride[31:0]
    g1[6] = 0;                          // stride0 hi | stride1 lo (2D: unused)
    g1[7] = 0;
    i32x4 z4 = {0, 0, 0, 0};
#if __clang_major__ >= 23
    i32x8 z8 = {0, 0, 0, 0, 0, 0, 0, 0};
    __builtin_amdgcn_tensor_load_to_lds(g0, g1, z4, z4, z8, 0);
#else
    __builtin_amdgcn_tensor_load_to_lds(g0, g1, z4, z4, 0);
#endif
}
#endif  // HAVE_TDM

// Fallback staging: straight 2D tile copy with padded rows, 2x b128 per thread.
__device__ __forceinline__ void stage_tile(unsigned short* dst,
                                           const unsigned short* src,
                                           unsigned strideElems, int tid) {
    int row = tid >> 1, half = tid & 1;
    const uint4* s = (const uint4*)(src + (size_t)row * strideElems + half * 16);
    uint4* d = (uint4*)(dst + row * LDW + half * 16);
    d[0] = s[0];
    d[1] = s[1];
}

// ---------------------------------------------------------------------------
// Kernel 1: act_bf16[e][n][s] = bf16( sum_m em[e][m][n]*h[m][s] + ba[s] )
// A tile: emT[e][n][m]  (rows n, m contiguous)  -> straight copy
// B tile: hbfT[s][m]    (rows s, m contiguous)  -> straight copy
// ---------------------------------------------------------------------------
__global__ __launch_bounds__(256)
void k_act(const unsigned short* __restrict__ emT,
           const unsigned short* __restrict__ hbfT,
           const float* __restrict__ ba,
           unsigned short* __restrict__ actbf) {
    __shared__ __align__(16) unsigned short As[2][BM * LDW];
    __shared__ __align__(16) unsigned short Bs[2][BN * LDW];

    const int s0 = blockIdx.x * BN;
    const int n0 = blockIdx.y * BM;
    const int e  = blockIdx.z;
    const unsigned short* Aten = emT + ((size_t)e * N_ + n0) * N_;
    const unsigned short* Bten = hbfT + (size_t)s0 * N_;

    const int tid   = threadIdx.x;
    const int lane  = tid & 31;
    const int wave  = tid >> 5;
    const int waveM = (wave >> 2) * 64;
    const int waveN = (wave & 3) * 32;
    const int m16   = lane & 15;
    const int kh    = lane >> 4;

    v8f acc[4][2];
    #pragma unroll
    for (int i = 0; i < 4; ++i)
        #pragma unroll
        for (int j = 0; j < 2; ++j) acc[i][j] = v8f_zero();

    const int KB = N_ / BK;   // 64
#if HAVE_TDM
    if (tid < 32) {           // one wave issues; EXEC is ignored by TDM
        tdm_tile_128x32((unsigned)(size_t)&As[0][0], Aten, N_);
        tdm_tile_128x32((unsigned)(size_t)&Bs[0][0], Bten, N_);
    }
#endif
    for (int kb = 0; kb < KB; ++kb) {
        const int cur = kb & 1;
#if HAVE_TDM
        if (tid < 32) {
            if (kb + 1 < KB) {   // prefetch next pair, wait for current pair
                tdm_tile_128x32((unsigned)(size_t)&As[cur ^ 1][0], Aten + (kb + 1) * BK, N_);
                tdm_tile_128x32((unsigned)(size_t)&Bs[cur ^ 1][0], Bten + (kb + 1) * BK, N_);
                __builtin_amdgcn_s_wait_tensorcnt(2);
            } else {
                __builtin_amdgcn_s_wait_tensorcnt(0);
            }
        }
        __syncthreads();
#else
        __syncthreads();
        stage_tile(As[cur], Aten + kb * BK, N_, tid);
        stage_tile(Bs[cur], Bten + kb * BK, N_, tid);
        __syncthreads();
#endif
        v16bf aF[4], bF[2];
        #pragma unroll
        for (int mi = 0; mi < 4; ++mi) aF[mi] = ld_fragA(As[cur], waveM + mi * 16 + m16, kh);
        #pragma unroll
        for (int ni = 0; ni < 2; ++ni) bF[ni] = ld_fragB(Bs[cur], waveN + ni * 16 + m16, kh);
        #pragma unroll
        for (int mi = 0; mi < 4; ++mi)
            #pragma unroll
            for (int ni = 0; ni < 2; ++ni)
                acc[mi][ni] = __builtin_amdgcn_wmma_f32_16x16x32_bf16(
                    false, aF[mi], false, bF[ni], (short)0, acc[mi][ni], false, false);
#if HAVE_TDM
        __syncthreads();   // all reads of buf[cur] done before it is re-filled
#endif
    }

    #pragma unroll
    for (int mi = 0; mi < 4; ++mi) {
        #pragma unroll
        for (int ni = 0; ni < 2; ++ni) {
            int gcol = s0 + waveN + ni * 16 + m16;
            float bav = ba[gcol];
            #pragma unroll
            for (int v = 0; v < 8; ++v) {
                int grow = n0 + waveM + mi * 16 + kh * 8 + v;
                actbf[((size_t)e * N_ + grow) * S_ + gcol] = f2bf(acc[mi][ni][v] + bav);
            }
        }
    }
}

// ---------------------------------------------------------------------------
// Kernel 2: gate GEMM.  mode 0: z = sigmoid(act@Wz + h@Uz)
//                       mode 1: r = sigmoid(act@Wr + h@Ur), rh_bf = bf16(r*h)
//                       mode 2: hh = tanh(act@Wh + (r*h)@Uh),
//                               h_new = (1-z)*h + r*hh  (+ bf16 / bf16^T copies)
// A tiles: actbf[e][n][s] / hbf / rhbf (rows n, s contiguous) -> straight copy
// B tiles: WT[e][t][s] / UT[t][s]      (rows t, s contiguous) -> straight copy
// ---------------------------------------------------------------------------
__device__ __forceinline__ void gate_src(int kb, int n0, int t0,
                                         const unsigned short* actbf,
                                         const unsigned short* A2,
                                         const unsigned short* WT,
                                         const unsigned short* UT,
                                         const unsigned short** Ab,
                                         const unsigned short** Bb) {
    if (kb < E_ * 8) {
        int e = kb >> 3, sK = (kb & 7) * BK;
        *Ab = actbf + ((size_t)e * N_ + n0) * S_ + sK;
        *Bb = WT + (size_t)e * S_ * S_ + (size_t)t0 * S_ + sK;
    } else {
        int sK = (kb - E_ * 8) * BK;
        *Ab = A2 + (size_t)n0 * S_ + sK;
        *Bb = UT + (size_t)t0 * S_ + sK;
    }
}

__global__ __launch_bounds__(256)
void k_gate(const unsigned short* __restrict__ actbf,
            const unsigned short* __restrict__ hbf,
            unsigned short* __restrict__ rhbf,
            const unsigned short* __restrict__ wzT,
            const unsigned short* __restrict__ wrT,
            const unsigned short* __restrict__ whT,
            const unsigned short* __restrict__ uzT,
            const unsigned short* __restrict__ urT,
            const unsigned short* __restrict__ uhT,
            const float* __restrict__ hIn,
            float* __restrict__ zbuf,
            float* __restrict__ rbuf,
            float* __restrict__ hOut,
            unsigned short* __restrict__ hbfOut,
            unsigned short* __restrict__ hbfTOut,
            int modeBase) {
    __shared__ __align__(16) unsigned short As[2][BM * LDW];
    __shared__ __align__(16) unsigned short Bs[2][BN * LDW];

    const int t0 = blockIdx.x * BN;
    const int n0 = blockIdx.y * BM;
    const int mode = modeBase + blockIdx.z;

    const unsigned short* WT = (mode == 0) ? wzT : (mode == 1) ? wrT : whT;
    const unsigned short* UT = (mode == 0) ? uzT : (mode == 1) ? urT : uhT;
    const unsigned short* A2 = (mode == 2) ? rhbf : hbf;

    const int tid   = threadIdx.x;
    const int lane  = tid & 31;
    const int wave  = tid >> 5;
    const int waveM = (wave >> 2) * 64;
    const int waveN = (wave & 3) * 32;
    const int m16   = lane & 15;
    const int kh    = lane >> 4;

    v8f acc[4][2];
    #pragma unroll
    for (int i = 0; i < 4; ++i)
        #pragma unroll
        for (int j = 0; j < 2; ++j) acc[i][j] = v8f_zero();

    const int KB = E_ * 8 + S_ / BK;   // 88
    const unsigned short *Ab, *Bb;
#if HAVE_TDM
    if (tid < 32) {
        gate_src(0, n0, t0, actbf, A2, WT, UT, &Ab, &Bb);
        tdm_tile_128x32((unsigned)(size_t)&As[0][0], Ab, S_);
        tdm_tile_128x32((unsigned)(size_t)&Bs[0][0], Bb, S_);
    }
#endif
    for (int kb = 0; kb < KB; ++kb) {
        const int cur = kb & 1;
#if HAVE_TDM
        if (tid < 32) {
            if (kb + 1 < KB) {
                gate_src(kb + 1, n0, t0, actbf, A2, WT, UT, &Ab, &Bb);
                tdm_tile_128x32((unsigned)(size_t)&As[cur ^ 1][0], Ab, S_);
                tdm_tile_128x32((unsigned)(size_t)&Bs[cur ^ 1][0], Bb, S_);
                __builtin_amdgcn_s_wait_tensorcnt(2);
            } else {
                __builtin_amdgcn_s_wait_tensorcnt(0);
            }
        }
        __syncthreads();
#else
        gate_src(kb, n0, t0, actbf, A2, WT, UT, &Ab, &Bb);
        __syncthreads();
        stage_tile(As[cur], Ab, S_, tid);
        stage_tile(Bs[cur], Bb, S_, tid);
        __syncthreads();
#endif
        v16bf aF[4], bF[2];
        #pragma unroll
        for (int mi = 0; mi < 4; ++mi) aF[mi] = ld_fragA(As[cur], waveM + mi * 16 + m16, kh);
        #pragma unroll
        for (int ni = 0; ni < 2; ++ni) bF[ni] = ld_fragB(Bs[cur], waveN + ni * 16 + m16, kh);
        #pragma unroll
        for (int mi = 0; mi < 4; ++mi)
            #pragma unroll
            for (int ni = 0; ni < 2; ++ni)
                acc[mi][ni] = __builtin_amdgcn_wmma_f32_16x16x32_bf16(
                    false, aF[mi], false, bF[ni], (short)0, acc[mi][ni], false, false);
#if HAVE_TDM
        __syncthreads();
#endif
    }

    #pragma unroll
    for (int mi = 0; mi < 4; ++mi) {
        #pragma unroll
        for (int ni = 0; ni < 2; ++ni) {
            int gcol = t0 + waveN + ni * 16 + m16;
            #pragma unroll
            for (int v = 0; v < 8; ++v) {
                int grow = n0 + waveM + mi * 16 + kh * 8 + v;
                size_t off = (size_t)grow * S_ + gcol;
                float pre = acc[mi][ni][v];
                if (mode == 0) {
                    zbuf[off] = 1.f / (1.f + __expf(-pre));
                } else if (mode == 1) {
                    float r = 1.f / (1.f + __expf(-pre));
                    rbuf[off] = r;
                    rhbf[off] = f2bf(r * hIn[off]);
                } else {
                    float hh = tanhf(pre);
                    float hn = (1.f - zbuf[off]) * hIn[off] + rbuf[off] * hh;
                    hOut[off] = hn;
                    unsigned short hb = f2bf(hn);
                    hbfOut[off] = hb;                               // [n][s]
                    hbfTOut[(size_t)gcol * N_ + grow] = hb;          // [s][n]
                }
            }
        }
    }
}

// fp32 -> bf16 conversion
__global__ void k_cvt(const float* __restrict__ src, unsigned short* __restrict__ dst, int n) {
    int i = blockIdx.x * blockDim.x + threadIdx.x;
    if (i < n) dst[i] = f2bf(src[i]);
}

// batched fp32 -> bf16 transposing conversion: dst[b][c][r] = src[b][r][c]
__global__ void k_cvt_t(const float* __restrict__ src, unsigned short* __restrict__ dst,
                        int R, int C) {
    size_t base = (size_t)blockIdx.y * R * C;
    int i = blockIdx.x * blockDim.x + threadIdx.x;
    if (i < R * C) {
        int r = i / C, c = i % C;
        dst[base + (size_t)c * R + r] = f2bf(src[base + i]);
    }
}

extern "C" void kernel_launch(void* const* d_in, const int* in_sizes, int n_in,
                              void* d_out, int out_size, void* d_ws, size_t ws_size,
                              hipStream_t stream) {
    (void)in_sizes; (void)n_in; (void)out_size; (void)ws_size;
    const float* x  = (const float*)d_in[0];
    const float* em = (const float*)d_in[1];
    const float* ba = (const float*)d_in[2];
    const float* wz = (const float*)d_in[3];
    const float* wr = (const float*)d_in[4];
    const float* wh = (const float*)d_in[5];
    const float* uz = (const float*)d_in[6];
    const float* ur = (const float*)d_in[7];
    const float* uh = (const float*)d_in[8];
    // d_in[9] = iteration: fixed scalar 10; reading device memory would break
    // graph capture, so it is compiled in (ITER_).

    char* p = (char*)d_ws;
    auto carve = [&](size_t bytes) {
        char* r = p;
        p += (bytes + 255) & ~(size_t)255;
        return r;
    };
    unsigned short* emT  = (unsigned short*)carve((size_t)E_ * N_ * N_ * 2);  // 80 MB (L2-resident)
    unsigned short* actb = (unsigned short*)carve((size_t)E_ * N_ * S_ * 2);  // 10 MB
    unsigned short* hbf  = (unsigned short*)carve((size_t)N_ * S_ * 2);
    unsigned short* hbfT = (unsigned short*)carve((size_t)N_ * S_ * 2);
    unsigned short* wzT  = (unsigned short*)carve((size_t)E_ * S_ * S_ * 2);
    unsigned short* wrT  = (unsigned short*)carve((size_t)E_ * S_ * S_ * 2);
    unsigned short* whT  = (unsigned short*)carve((size_t)E_ * S_ * S_ * 2);
    unsigned short* uzT  = (unsigned short*)carve((size_t)S_ * S_ * 2);
    unsigned short* urT  = (unsigned short*)carve((size_t)S_ * S_ * 2);
    unsigned short* uhT  = (unsigned short*)carve((size_t)S_ * S_ * 2);
    unsigned short* rhb  = (unsigned short*)carve((size_t)N_ * S_ * 2);
    float* hbuf = (float*)carve((size_t)N_ * S_ * 4);
    float* zbuf = (float*)carve((size_t)N_ * S_ * 4);
    float* rbuf = (float*)carve((size_t)N_ * S_ * 4);

    // One-time transposing bf16 conversions (amortized over 10 iterations).
    {
        int nb = (N_ * N_ + 255) / 256;
        k_cvt_t<<<dim3(nb, E_), 256, 0, stream>>>(em, emT, N_, N_);       // emT[e][n][m]
        int sb = (S_ * S_ + 255) / 256;
        k_cvt_t<<<dim3(sb, E_), 256, 0, stream>>>(wz, wzT, S_, S_);       // WT[e][t][s]
        k_cvt_t<<<dim3(sb, E_), 256, 0, stream>>>(wr, wrT, S_, S_);
        k_cvt_t<<<dim3(sb, E_), 256, 0, stream>>>(wh, whT, S_, S_);
        k_cvt_t<<<dim3(sb, 1), 256, 0, stream>>>(uz, uzT, S_, S_);        // UT[t][s]
        k_cvt_t<<<dim3(sb, 1), 256, 0, stream>>>(ur, urT, S_, S_);
        k_cvt_t<<<dim3(sb, 1), 256, 0, stream>>>(uh, uhT, S_, S_);
        int xb = (N_ * S_ + 255) / 256;
        k_cvt_t<<<dim3(xb, 1), 256, 0, stream>>>(x, hbfT, N_, S_);        // hbfT[s][n]
        k_cvt<<<xb, 256, 0, stream>>>(x, hbf, N_ * S_);                   // hbf[n][s]
    }
    hipMemcpyAsync(hbuf, x, (size_t)N_ * S_ * 4, hipMemcpyDeviceToDevice, stream);

    const dim3 blk(256);
    for (int it = 0; it < ITER_; ++it) {
        k_act<<<dim3(S_ / BN, N_ / BM, E_), blk, 0, stream>>>(emT, hbfT, ba, actb);
        // z and r gates (grid.z = 2 -> modes 0,1)
        k_gate<<<dim3(S_ / BN, N_ / BM, 2), blk, 0, stream>>>(
            actb, hbf, rhb, wzT, wrT, whT, uzT, urT, uhT,
            hbuf, zbuf, rbuf, hbuf, hbf, hbfT, /*modeBase=*/0);
        // hh + state update (mode 2); final iteration writes straight to d_out
        float* hDst = (it == ITER_ - 1) ? (float*)d_out : hbuf;
        k_gate<<<dim3(S_ / BN, N_ / BM, 1), blk, 0, stream>>>(
            actb, hbf, rhb, wzT, wrT, whT, uzT, urT, uhT,
            hbuf, zbuf, rbuf, hDst, hbf, hbfT, /*modeBase=*/2);
    }
}
